// nri_rec_decoder_32049045962804
// MI455X (gfx1250) — matrix operand order repaired and checked by reading, not compiled, not run on hardware
//
#include <hip/hip_runtime.h>
#include <hip/hip_bf16.h>
#include <stdint.h>
#include <math.h>

typedef __attribute__((ext_vector_type(16))) __bf16 v16bf;
typedef __attribute__((ext_vector_type(8)))  float  v8f;
typedef __attribute__((ext_vector_type(8)))  __bf16 bf16x8;
typedef __attribute__((ext_vector_type(4)))  __bf16 bf16x4;

#define NN   10000
#define EE   160000
#define TT   64
#define NINP 64
#define NHID 128

// ---------------- setup kernels ----------------

__global__ void k_zero32(uint32_t* p, int n) {
  int i = blockIdx.x * blockDim.x + threadIdx.x;
  if (i < n) p[i] = 0u;
}

__global__ void k_count(const int* __restrict__ dst, int* __restrict__ deg) {
  int e = blockIdx.x * blockDim.x + threadIdx.x;
  if (e >= EE + NN) return;
  int d = (e < EE) ? dst[e] : (e - EE);
  atomicAdd(&deg[d], 1);
}

__global__ void k_dinv(const int* __restrict__ deg, float* __restrict__ dinv) {
  int n = blockIdx.x * blockDim.x + threadIdx.x;
  if (n < NN) dinv[n] = rsqrtf((float)deg[n]);   // deg >= 1 (self-loop)
}

// single-block exclusive prefix sum over deg -> offs[0..NN]
__global__ void k_scan(const int* __restrict__ deg, int* __restrict__ offs) {
  __shared__ int sm[1024];
  __shared__ int carry_s;
  if (threadIdx.x == 0) carry_s = 0;
  __syncthreads();
  for (int base = 0; base < NN; base += 1024) {
    int i = base + (int)threadIdx.x;
    int v = (i < NN) ? deg[i] : 0;
    sm[threadIdx.x] = v;
    __syncthreads();
    for (int off = 1; off < 1024; off <<= 1) {
      int t = (threadIdx.x >= (unsigned)off) ? sm[threadIdx.x - off] : 0;
      __syncthreads();
      sm[threadIdx.x] += t;
      __syncthreads();
    }
    int incl = sm[threadIdx.x];
    int c = carry_s;
    if (i < NN) offs[i] = c + incl - v;   // exclusive
    __syncthreads();
    if (threadIdx.x == 1023) carry_s = c + sm[1023];
    __syncthreads();
  }
  if (threadIdx.x == 0) offs[NN] = carry_s;   // = EE+NN
}

__global__ void k_initcur(const int* __restrict__ offs, int* __restrict__ cursor,
                          int* __restrict__ cursor2) {
  int n = blockIdx.x * blockDim.x + threadIdx.x;
  if (n < NN) { cursor[n] = offs[n]; cursor2[n] = offs[n] - n; }
}

// counting-sort fill: bucket edges (incl. self-loops) by destination
__global__ void k_fill(const int* __restrict__ src, const int* __restrict__ dst,
                       const float* __restrict__ dinv, int* __restrict__ cursor,
                       int* __restrict__ cursor2, int* __restrict__ srt_src,
                       float* __restrict__ srt_norm, int* __restrict__ srt2_eid) {
  int e = blockIdx.x * blockDim.x + threadIdx.x;
  if (e >= EE + NN) return;
  int s = (e < EE) ? src[e] : (e - EE);
  int d = (e < EE) ? dst[e] : (e - EE);
  float w = dinv[s] * dinv[d];
  int pos = atomicAdd(&cursor[d], 1);
  srt_src[pos] = s;
  srt_norm[pos] = w;
  if (e < EE) {                       // dst-only buckets (no self loops), edge ids
    int pos2 = atomicAdd(&cursor2[d], 1);
    srt2_eid[pos2] = e;
  }
}

// swizzle a [out,K] f32 weight matrix into WMMA B-fragment order:
// frag[ct][kt][lane][e] = bf16( W[(ct*16 + (lane&15)) * K + kt*32 + (e>>3)*16 + (lane>>4)*8 + (e&7)] )
__global__ void k_prep_frag(const float* __restrict__ W, __bf16* __restrict__ frag,
                            int ncol_tiles, int nk_tiles, int K) {
  int tid = blockIdx.x * blockDim.x + threadIdx.x;
  int total = ncol_tiles * nk_tiles * 32;
  if (tid >= total) return;
  int lane = tid & 31;
  int t = tid >> 5;
  int kt = t % nk_tiles;
  int ct = t / nk_tiles;
  int n = lane & 15, khalf = lane >> 4;
  int col = ct * 16 + n;
#pragma unroll
  for (int e = 0; e < 16; ++e) {
    int k = kt * 32 + (e >> 3) * 16 + khalf * 8 + (e & 7);
    frag[(size_t)tid * 16 + e] = (__bf16)W[(size_t)col * K + k];
  }
}

__global__ void k_bias512(const float* bi, const float* bfv, const float* bo,
                          const float* bg, float* bias512) {
  int j = blockIdx.x * blockDim.x + threadIdx.x;
  if (j >= 512) return;
  float v;
  if (j < 128)      v = bi[j];
  else if (j < 256) v = bfv[j - 128];
  else if (j < 384) v = bo[j - 256];
  else              v = bg[j - 384];
  bias512[j] = v;
}

// ---------------- per-step kernels ----------------

// load 8 contiguous "comb" values (xt | h) as bf16, per ISA A-fragment layout
__device__ __forceinline__ void load_comb8(const float* __restrict__ xt,
                                           const __bf16* __restrict__ hbf,
                                           int row, int k, __bf16* out) {
  if (k < NINP) {
    const float4* p = (const float4*)(xt + (size_t)row * NINP + k);
    float4 a = p[0], b = p[1];
    out[0] = (__bf16)a.x; out[1] = (__bf16)a.y; out[2] = (__bf16)a.z; out[3] = (__bf16)a.w;
    out[4] = (__bf16)b.x; out[5] = (__bf16)b.y; out[6] = (__bf16)b.z; out[7] = (__bf16)b.w;
  } else {
    bf16x8 v = *(const bf16x8*)(hbf + (size_t)row * NHID + (k - NINP));
#pragma unroll
    for (int i = 0; i < 8; ++i) out[i] = v[i];
  }
}

// zw[N,512] = [xt | h] @ [Wi;Wf;Wo;Wg]^T  via bf16 WMMA.  grid = N/16, block 256.
__global__ void __launch_bounds__(256) k_gemm_gates(
    const float* __restrict__ xt, const __bf16* __restrict__ hbf,
    const __bf16* __restrict__ Wfrag, float* __restrict__ zw) {
  int lane = threadIdx.x & 31;
  int wave = threadIdx.x >> 5;
  int row0 = blockIdx.x * 16;
  int m = lane & 15, half = lane >> 4;
  int row = row0 + m;
  int ct0 = wave * 4;                 // 8 waves x 4 col-tiles = 32 tiles = 512 cols

  v8f acc[4] = {};
#pragma unroll
  for (int kt = 0; kt < 6; ++kt) {    // K = 192 = 6 x 32
    int kb = kt * 32;
    v16bf a;
    __bf16 tmp[8];
    load_comb8(xt, hbf, row, kb + half * 8, tmp);
#pragma unroll
    for (int i = 0; i < 8; ++i) a[i] = tmp[i];
    load_comb8(xt, hbf, row, kb + 16 + half * 8, tmp);
#pragma unroll
    for (int i = 0; i < 8; ++i) a[8 + i] = tmp[i];

#pragma unroll
    for (int c = 0; c < 4; ++c) {
      int ct = ct0 + c;
      const bf16x8* bp = (const bf16x8*)(Wfrag + (((size_t)ct * 6 + kt) * 32 + lane) * 16);
      bf16x8 b0 = bp[0], b1 = bp[1];
      v16bf bfrag;
#pragma unroll
      for (int i = 0; i < 8; ++i) { bfrag[i] = b0[i]; bfrag[8 + i] = b1[i]; }
      acc[c] = __builtin_amdgcn_wmma_f32_16x16x32_bf16(false, a, false, bfrag,
                                                       (short)0, acc[c], false, false);
    }
  }
#pragma unroll
  for (int c = 0; c < 4; ++c) {
    int colb = (ct0 + c) * 16 + m;    // D: N = lane&15
#pragma unroll
    for (int r = 0; r < 8; ++r) {     // D: M = r + (lane>>4)*8
      int gr = row0 + half * 8 + r;
      zw[(size_t)gr * 512 + colb] = acc[c][r];
    }
  }
}

// gates[n,512] = sum_{edges -> n} norm * zw[src] + bias.  One wave per node.
__global__ void __launch_bounds__(256) k_gather_gates(
    const float* __restrict__ zw, const int* __restrict__ srt_src,
    const float* __restrict__ srt_norm, const int* __restrict__ offs,
    const float* __restrict__ bias512, float* __restrict__ gates) {
  int lane = threadIdx.x & 31;
  int wave = threadIdx.x >> 5;
  int n = blockIdx.x * 8 + wave;
  if (n >= NN) return;
  int j0 = offs[n], j1 = offs[n + 1];
  float4 a0 = {0,0,0,0}, a1 = {0,0,0,0}, a2 = {0,0,0,0}, a3 = {0,0,0,0};
  for (int j = j0; j < j1; ++j) {
    int s = srt_src[j];
    float w = srt_norm[j];
    const float4* zp = (const float4*)(zw + (size_t)s * 512);
    float4 v0 = zp[lane], v1 = zp[32 + lane], v2 = zp[64 + lane], v3 = zp[96 + lane];
    a0.x += w*v0.x; a0.y += w*v0.y; a0.z += w*v0.z; a0.w += w*v0.w;
    a1.x += w*v1.x; a1.y += w*v1.y; a1.z += w*v1.z; a1.w += w*v1.w;
    a2.x += w*v2.x; a2.y += w*v2.y; a2.z += w*v2.z; a2.w += w*v2.w;
    a3.x += w*v3.x; a3.y += w*v3.y; a3.z += w*v3.z; a3.w += w*v3.w;
  }
  const float4* bp = (const float4*)bias512;
  float4 b0 = bp[lane], b1 = bp[32 + lane], b2 = bp[64 + lane], b3 = bp[96 + lane];
  a0.x += b0.x; a0.y += b0.y; a0.z += b0.z; a0.w += b0.w;
  a1.x += b1.x; a1.y += b1.y; a1.z += b1.z; a1.w += b1.w;
  a2.x += b2.x; a2.y += b2.y; a2.z += b2.z; a2.w += b2.w;
  a3.x += b3.x; a3.y += b3.y; a3.z += b3.z; a3.w += b3.w;
  float4* gp = (float4*)(gates + (size_t)n * 512);
  gp[lane] = a0; gp[32 + lane] = a1; gp[64 + lane] = a2; gp[96 + lane] = a3;
}

__global__ void k_lstm(const float* __restrict__ gates, float* __restrict__ c,
                       __bf16* __restrict__ hbf) {
  int idx = blockIdx.x * blockDim.x + threadIdx.x;
  if (idx >= NN * NHID) return;
  int n = idx >> 7, j = idx & 127;
  const float* g = gates + (size_t)n * 512;
  float iv = 1.f / (1.f + __expf(-g[j]));
  float fv = 1.f / (1.f + __expf(-g[128 + j]));
  float ov = 1.f / (1.f + __expf(-g[256 + j]));
  float gv = tanhf(g[384 + j]);
  float cv = fv * c[idx] + iv * gv;
  c[idx] = cv;
  hbf[idx] = (__bf16)(ov * tanhf(cv));
}

// ---------------- post-scan kernels ----------------

// ebf[E,128] = relu( [h[m_in] | h[m_out]] @ W1^T + b1 )  (bf16 out)
__global__ void __launch_bounds__(256) k_gemm_edge(
    const __bf16* __restrict__ hbf, const int* __restrict__ m_in,
    const int* __restrict__ m_out, const __bf16* __restrict__ W1frag,
    const float* __restrict__ b1, __bf16* __restrict__ ebf) {
  int lane = threadIdx.x & 31;
  int wave = threadIdx.x >> 5;
  int rt = blockIdx.x * 2 + (wave >> 2);
  int row0 = rt * 16;
  int m = lane & 15, half = lane >> 4;
  int row = row0 + m;
  int nin = m_in[row], nout = m_out[row];
  int ct0 = (wave & 3) * 2;

  v8f acc[2] = {};
#pragma unroll
  for (int kt = 0; kt < 8; ++kt) {    // K = 256 = 8 x 32 (kt<4: h[m_in], kt>=4: h[m_out])
    int k0 = kt * 32 + half * 8;
    int k1 = kt * 32 + 16 + half * 8;
    const __bf16* p0 = (k0 < 128) ? (hbf + (size_t)nin * 128 + k0)
                                  : (hbf + (size_t)nout * 128 + (k0 - 128));
    const __bf16* p1 = (k1 < 128) ? (hbf + (size_t)nin * 128 + k1)
                                  : (hbf + (size_t)nout * 128 + (k1 - 128));
    bf16x8 va = *(const bf16x8*)p0;
    bf16x8 vb = *(const bf16x8*)p1;
    v16bf a;
#pragma unroll
    for (int i = 0; i < 8; ++i) { a[i] = va[i]; a[8 + i] = vb[i]; }
#pragma unroll
    for (int cc = 0; cc < 2; ++cc) {
      int ct = ct0 + cc;
      const bf16x8* bp = (const bf16x8*)(W1frag + (((size_t)ct * 8 + kt) * 32 + lane) * 16);
      bf16x8 b0 = bp[0], b1v = bp[1];
      v16bf bfrag;
#pragma unroll
      for (int i = 0; i < 8; ++i) { bfrag[i] = b0[i]; bfrag[8 + i] = b1v[i]; }
      acc[cc] = __builtin_amdgcn_wmma_f32_16x16x32_bf16(false, a, false, bfrag,
                                                        (short)0, acc[cc], false, false);
    }
  }
#pragma unroll
  for (int cc = 0; cc < 2; ++cc) {
    int col = (ct0 + cc) * 16 + m;
    float bias = b1[col];
#pragma unroll
    for (int r = 0; r < 8; ++r) {
      int er = row0 + half * 8 + r;
      float v = acc[cc][r] + bias;
      v = v > 0.f ? v : 0.f;
      ebf[(size_t)er * 128 + col] = (__bf16)v;
    }
  }
}

// nodes_bf[n,128] = sum over dst-only edges of ebf[eid]
__global__ void __launch_bounds__(256) k_gather_nodes(
    const __bf16* __restrict__ ebf, const int* __restrict__ srt2_eid,
    const int* __restrict__ offs, __bf16* __restrict__ nodes_bf) {
  int lane = threadIdx.x & 31;
  int wave = threadIdx.x >> 5;
  int n = blockIdx.x * 8 + wave;
  if (n >= NN) return;
  int j0 = offs[n] - n, j1 = offs[n + 1] - (n + 1);   // dst-only bucket bounds
  float acc[4] = {0.f, 0.f, 0.f, 0.f};
  for (int j = j0; j < j1; ++j) {
    int e = srt2_eid[j];
    bf16x4 v = *(const bf16x4*)(ebf + (size_t)e * 128 + lane * 4);
#pragma unroll
    for (int i = 0; i < 4; ++i) acc[i] += (float)v[i];
  }
#pragma unroll
  for (int i = 0; i < 4; ++i)
    nodes_bf[(size_t)n * 128 + lane * 4 + i] = (__bf16)acc[i];
}

// zwc[N,64] = nodes_bf @ Wc^T
__global__ void __launch_bounds__(128) k_gemm_out(
    const __bf16* __restrict__ nodes_bf, const __bf16* __restrict__ Wcfrag,
    float* __restrict__ zwc) {
  int lane = threadIdx.x & 31;
  int ct = threadIdx.x >> 5;          // 4 waves = 4 col tiles = 64 cols
  int row0 = blockIdx.x * 16;
  int m = lane & 15, half = lane >> 4;
  int row = row0 + m;

  v8f acc = {};
#pragma unroll
  for (int kt = 0; kt < 4; ++kt) {    // K = 128
    bf16x8 va = *(const bf16x8*)(nodes_bf + (size_t)row * 128 + kt * 32 + half * 8);
    bf16x8 vb = *(const bf16x8*)(nodes_bf + (size_t)row * 128 + kt * 32 + 16 + half * 8);
    v16bf a;
#pragma unroll
    for (int i = 0; i < 8; ++i) { a[i] = va[i]; a[8 + i] = vb[i]; }
    const bf16x8* bp = (const bf16x8*)(Wcfrag + (((size_t)ct * 4 + kt) * 32 + lane) * 16);
    bf16x8 b0 = bp[0], b1 = bp[1];
    v16bf bfrag;
#pragma unroll
    for (int i = 0; i < 8; ++i) { bfrag[i] = b0[i]; bfrag[8 + i] = b1[i]; }
    acc = __builtin_amdgcn_wmma_f32_16x16x32_bf16(false, a, false, bfrag,
                                                  (short)0, acc, false, false);
  }
  int col = ct * 16 + m;
#pragma unroll
  for (int r = 0; r < 8; ++r)
    zwc[(size_t)(row0 + half * 8 + r) * 64 + col] = acc[r];
}

// out[n,64] = sum norm * zwc[src] + bc
__global__ void __launch_bounds__(256) k_gather_out(
    const float* __restrict__ zwc, const int* __restrict__ srt_src,
    const float* __restrict__ srt_norm, const int* __restrict__ offs,
    const float* __restrict__ bc, float* __restrict__ out) {
  int lane = threadIdx.x & 31;
  int wave = threadIdx.x >> 5;
  int n = blockIdx.x * 8 + wave;
  if (n >= NN) return;
  int j0 = offs[n], j1 = offs[n + 1];
  float2 acc = {0.f, 0.f};
  for (int j = j0; j < j1; ++j) {
    int s = srt_src[j];
    float w = srt_norm[j];
    float2 v = ((const float2*)(zwc + (size_t)s * 64))[lane];
    acc.x += w * v.x; acc.y += w * v.y;
  }
  float2 b = ((const float2*)bc)[lane];
  acc.x += b.x; acc.y += b.y;
  ((float2*)(out + (size_t)n * 64))[lane] = acc;
}

// ---------------- host launcher ----------------

extern "C" void kernel_launch(void* const* d_in, const int* in_sizes, int n_in,
                              void* d_out, int out_size, void* d_ws, size_t ws_size,
                              hipStream_t stream) {
  (void)in_sizes; (void)n_in; (void)out_size; (void)ws_size;
  const float* x    = (const float*)d_in[0];
  const int*   m_in = (const int*)d_in[2];    // dst
  const int*   m_out= (const int*)d_in[3];    // src
  const float* Wi = (const float*)d_in[4];  const float* bi  = (const float*)d_in[5];
  const float* Wf = (const float*)d_in[6];  const float* bfv = (const float*)d_in[7];
  const float* Wo = (const float*)d_in[8];  const float* bo  = (const float*)d_in[9];
  const float* Wg = (const float*)d_in[10]; const float* bg  = (const float*)d_in[11];
  const float* W1 = (const float*)d_in[12]; const float* b1  = (const float*)d_in[13];
  const float* Wc = (const float*)d_in[14]; const float* bc  = (const float*)d_in[15];
  float* out = (float*)d_out;

  char* p = (char*)d_ws;
  auto alloc = [&](size_t bytes) -> char* {
    char* r = p; p += (bytes + 255) & ~(size_t)255; return r;
  };
  int*    deg      = (int*)alloc((size_t)NN * 4);
  int*    offs     = (int*)alloc((size_t)(NN + 1) * 4);
  int*    cursor   = (int*)alloc((size_t)NN * 4);
  int*    cursor2  = (int*)alloc((size_t)NN * 4);
  float*  dinv     = (float*)alloc((size_t)NN * 4);
  int*    srt_src  = (int*)alloc((size_t)(EE + NN) * 4);
  float*  srt_norm = (float*)alloc((size_t)(EE + NN) * 4);
  int*    srt2     = (int*)alloc((size_t)EE * 4);
  __bf16* Wallfrag = (__bf16*)alloc((size_t)512 * 192 * 2);
  __bf16* W1frag   = (__bf16*)alloc((size_t)128 * 256 * 2);
  __bf16* Wcfrag   = (__bf16*)alloc((size_t)64 * 128 * 2);
  float*  bias512  = (float*)alloc(512 * 4);
  __bf16* hbf      = (__bf16*)alloc((size_t)NN * 128 * 2);
  float*  cst      = (float*)alloc((size_t)NN * 128 * 4);
  float*  zw       = (float*)alloc((size_t)NN * 512 * 4);   // 20.48 MB
  float*  gates    = (float*)alloc((size_t)NN * 512 * 4);   // 20.48 MB
  __bf16* ebf      = (__bf16*)zw;   // alias E*128 bf16 = 40.96 MB over zw+gates (dead post-loop)
  __bf16* nodes_bf = (__bf16*)alloc((size_t)NN * 128 * 2);
  float*  zwc      = (float*)alloc((size_t)NN * 64 * 4);
  (void)gates;

  // ---- setup ----
  k_zero32<<<(NN + 255) / 256, 256, 0, stream>>>((uint32_t*)deg, NN);
  k_zero32<<<(NN * 128 / 2 + 255) / 256, 256, 0, stream>>>((uint32_t*)hbf, NN * 128 / 2);
  k_zero32<<<(NN * 128 + 255) / 256, 256, 0, stream>>>((uint32_t*)cst, NN * 128);
  k_count<<<(EE + NN + 255) / 256, 256, 0, stream>>>(m_in, deg);
  k_dinv<<<(NN + 255) / 256, 256, 0, stream>>>(deg, dinv);
  k_scan<<<1, 1024, 0, stream>>>(deg, offs);
  k_initcur<<<(NN + 255) / 256, 256, 0, stream>>>(offs, cursor, cursor2);
  k_fill<<<(EE + NN + 255) / 256, 256, 0, stream>>>(m_out, m_in, dinv, cursor, cursor2,
                                                    srt_src, srt_norm, srt2);
  // weight fragments: Wall = [Wi;Wf;Wo;Wg], each gate = 8 col-tiles x 6 k-tiles
  {
    int ft = 8 * 6 * 32;
    k_prep_frag<<<(ft + 255) / 256, 256, 0, stream>>>(Wi, Wallfrag + (size_t)0 * ft * 16, 8, 6, 192);
    k_prep_frag<<<(ft + 255) / 256, 256, 0, stream>>>(Wf, Wallfrag + (size_t)1 * ft * 16, 8, 6, 192);
    k_prep_frag<<<(ft + 255) / 256, 256, 0, stream>>>(Wo, Wallfrag + (size_t)2 * ft * 16, 8, 6, 192);
    k_prep_frag<<<(ft + 255) / 256, 256, 0, stream>>>(Wg, Wallfrag + (size_t)3 * ft * 16, 8, 6, 192);
  }
  k_prep_frag<<<(8 * 8 * 32 + 255) / 256, 256, 0, stream>>>(W1, W1frag, 8, 8, 256);
  k_prep_frag<<<(4 * 4 * 32 + 255) / 256, 256, 0, stream>>>(Wc, Wcfrag, 4, 4, 128);
  k_bias512<<<2, 256, 0, stream>>>(bi, bfv, bo, bg, bias512);

  // ---- recurrent scan ----
  for (int t = 0; t < TT; ++t) {
    const float* xt = x + (size_t)t * NN * NINP;
    k_gemm_gates<<<NN / 16, 256, 0, stream>>>(xt, hbf, Wallfrag, zw);
    k_gather_gates<<<(NN + 7) / 8, 256, 0, stream>>>(zw, srt_src, srt_norm, offs, bias512, gates);
    k_lstm<<<(NN * 128 + 255) / 256, 256, 0, stream>>>(gates, cst, hbf);
  }

  // ---- decoder tail ----
  k_gemm_edge<<<EE / 32, 256, 0, stream>>>(hbf, m_in, m_out, W1frag, b1, ebf);
  k_gather_nodes<<<(NN + 7) / 8, 256, 0, stream>>>(ebf, srt2, offs, nodes_bf);
  k_gemm_out<<<NN / 16, 128, 0, stream>>>(nodes_bf, Wcfrag, zwc);
  k_gather_out<<<(NN + 7) / 8, 256, 0, stream>>>(zwc, srt_src, srt_norm, offs, bc, out);
}
